// HimpNetAlternative_68049461838548
// MI455X (gfx1250) — matrix operand order repaired and checked by reading, not compiled, not run on hardware
//
#include <hip/hip_runtime.h>

// ---------------------------------------------------------------------------
// Fused edge-MLP (2x [E,256]x[256,256] bf16 WMMA GEMM + bias + ReLU) followed
// by scatter-mean into N nodes.  Memory-bound: stream X once, keep weights &
// partial sums in L2, keep the layer-1 activation in LDS (never touches HBM).
//
// Round-3 refinements (from gfx1250 asm inspection):
//  * 4 independent WMMA accumulator chains per wave (shared A fragments)
//  * source-level double buffering of B fragments -> partial dscnt waits
//  * float4-vectorized global staging, packed bf16x4 LDS stores
//  * node indices hoisted out of the layer-2 tile loop
// ---------------------------------------------------------------------------

#define EDGES      320000
#define NODES      10000
#define CH         256          // in/out channels
#define TE         128          // edges per block
#define RS         264          // padded LDS row stride (bf16 elems), 528B rows
#define NTHREADS   256          // 8 waves (wave32)

typedef __attribute__((ext_vector_type(16))) __bf16 v16bf;
typedef __attribute__((ext_vector_type(8)))  float  v8f;

union FragQ {
    uint4 q[2];
    v16bf v;
};

__device__ __forceinline__ __bf16 to_bf16(float f) { return (__bf16)f; }

__device__ __forceinline__ void store_bf16x4(__bf16* p, float4 v) {
    union { __bf16 h[4]; uint2 u; } t;
    t.h[0] = (__bf16)v.x; t.h[1] = (__bf16)v.y;
    t.h[2] = (__bf16)v.z; t.h[3] = (__bf16)v.w;
    *(uint2*)p = t.u;
}

// A fragment: 16-bit A 16x32 layout. Lane (half = lane>>4, M = lane&15).
// elems 0..7  <-> K = k0 + 8*half + e         (16 contiguous bytes)
// elems 8..15 <-> K = k0 + 8*half + 16 + (e-8) (16 contiguous bytes)
__device__ __forceinline__ FragQ load_a(const __bf16* base, int row, int k0, int half) {
    const uint4* p = (const uint4*)(base + row * RS + k0 + 8 * half);
    FragQ f;
    f.q[0] = p[0];
    f.q[1] = p[2];   // +16 elements = +32 bytes
    return f;
}

// B fragment: 16-bit B 32x16 layout. Lane n = lane&15.
// elems 0..7  <-> K = k0 + 8*half + e; elems 8..15 <-> K = k0 + 8*half + 16 + (e-8)
__device__ __forceinline__ FragQ load_b(const __bf16* base, int row, int k0, int half) {
    const uint4* p = (const uint4*)(base + row * RS + k0 + 8 * half);
    FragQ f;
    f.q[0] = p[0];
    f.q[1] = p[2];
    return f;
}

// ------------------------- zero workspace ----------------------------------
__global__ void zero_ws_kernel(float* sums, float* cnt) {
    int i = blockIdx.x * blockDim.x + threadIdx.x;
    if (i < NODES * CH) sums[i] = 0.0f;
    if (i < NODES)      cnt[i]  = 0.0f;
}

// ------------------------- per-node edge counts ----------------------------
__global__ void count_kernel(const int* __restrict__ idx, float* cnt) {
    int i = blockIdx.x * blockDim.x + threadIdx.x;
    if (i < EDGES) atomicAdd(&cnt[idx[i]], 1.0f);
}

// ------------------------- fused MLP + scatter-add -------------------------
__global__ __launch_bounds__(NTHREADS)
void mlp_scatter_kernel(const float* __restrict__ X,
                        const int*   __restrict__ idx,
                        const float* __restrict__ W1,
                        const float* __restrict__ b1,
                        const float* __restrict__ W2,
                        const float* __restrict__ b2,
                        float* __restrict__ sums) {
    extern __shared__ __bf16 sm[];
    __bf16* Xt = sm;                    // [TE][RS]  bf16 edge-feature tile
    __bf16* Wt = sm + TE * RS;          // [CH][RS]  bf16 transposed weights (W^T[n][k])
    __bf16* M1 = Wt + CH * RS;          // [TE][RS]  bf16 layer-1 activation

    const int tid  = threadIdx.x;
    const int e0   = blockIdx.x * TE;
    const int lane = tid & 31;
    const int wv   = tid >> 5;          // wave id 0..7 -> fixed M-tile
    const int half = lane >> 4;
    const int ln   = lane & 15;
    const int eb   = wv * 16;           // this wave's 16-edge strip

    // ---- stage X tile (float4 loads -> packed bf16x4 stores) ----
    {
        const float4* X4 = (const float4*)X;
        for (int i = tid; i < TE * (CH / 4); i += NTHREADS) {
            int e = i >> 6;             // 64 float4 per row
            int c4 = i & 63;
            float4 x = X4[(size_t)(e0 + e) * (CH / 4) + c4];
            store_bf16x4(&Xt[e * RS + c4 * 4], x);
        }
    }
    // ---- stage W1^T (coalesced float4 reads, scattered b16 LDS writes) ----
    {
        const float4* W4 = (const float4*)W1;
        for (int i = tid; i < CH * (CH / 4); i += NTHREADS) {
            int k = i >> 6;
            int n4 = (i & 63) * 4;
            float4 w = W4[i];
            Wt[(n4 + 0) * RS + k] = to_bf16(w.x);
            Wt[(n4 + 1) * RS + k] = to_bf16(w.y);
            Wt[(n4 + 2) * RS + k] = to_bf16(w.z);
            Wt[(n4 + 3) * RS + k] = to_bf16(w.w);
        }
    }
    __syncthreads();

    // ================= layer 1: M1 = relu(X @ W1 + b1) =================
    {
        FragQ a[8];
#pragma unroll
        for (int kk = 0; kk < 8; ++kk)
            a[kk] = load_a(Xt, eb + ln, kk * 32, half);

        for (int t = 0; t < 4; ++t) {
            int nb[4];
#pragma unroll
            for (int c = 0; c < 4; ++c) nb[c] = t * 16 + c * 64;

            v8f acc[4] = {{}, {}, {}, {}};
            FragQ bc[4];
#pragma unroll
            for (int c = 0; c < 4; ++c) bc[c] = load_b(Wt, nb[c] + ln, 0, half);
#pragma unroll
            for (int kk = 0; kk < 7; ++kk) {
                FragQ bn[4];
#pragma unroll
                for (int c = 0; c < 4; ++c)
                    bn[c] = load_b(Wt, nb[c] + ln, (kk + 1) * 32, half);
#pragma unroll
                for (int c = 0; c < 4; ++c)
                    acc[c] = __builtin_amdgcn_wmma_f32_16x16x32_bf16(
                        false, a[kk].v, false, bc[c].v, (short)0, acc[c], false, false);
#pragma unroll
                for (int c = 0; c < 4; ++c) bc[c] = bn[c];
            }
#pragma unroll
            for (int c = 0; c < 4; ++c)
                acc[c] = __builtin_amdgcn_wmma_f32_16x16x32_bf16(
                    false, a[7].v, false, bc[c].v, (short)0, acc[c], false, false);

#pragma unroll
            for (int c = 0; c < 4; ++c) {
                float bias = b1[nb[c] + ln];
#pragma unroll
                for (int v = 0; v < 8; ++v) {
                    int M = v + 8 * half;
                    M1[(eb + M) * RS + (nb[c] + ln)] =
                        to_bf16(fmaxf(acc[c][v] + bias, 0.0f));
                }
            }
        }
    }
    __syncthreads();

    // ---- swap in W2^T ----
    {
        const float4* W4 = (const float4*)W2;
        for (int i = tid; i < CH * (CH / 4); i += NTHREADS) {
            int k = i >> 6;
            int n4 = (i & 63) * 4;
            float4 w = W4[i];
            Wt[(n4 + 0) * RS + k] = to_bf16(w.x);
            Wt[(n4 + 1) * RS + k] = to_bf16(w.y);
            Wt[(n4 + 2) * RS + k] = to_bf16(w.z);
            Wt[(n4 + 3) * RS + k] = to_bf16(w.w);
        }
    }
    __syncthreads();

    // ======== layer 2: out = relu(M1 @ W2 + b2) -> atomic scatter ========
    {
        FragQ a[8];
#pragma unroll
        for (int kk = 0; kk < 8; ++kk)
            a[kk] = load_a(M1, eb + ln, kk * 32, half);

        // row offsets into sums[] for this wave's 16 edges (t-invariant)
        int noff[8];
#pragma unroll
        for (int v = 0; v < 8; ++v)
            noff[v] = idx[e0 + eb + v + 8 * half] * CH;

        for (int t = 0; t < 4; ++t) {
            int nb[4];
#pragma unroll
            for (int c = 0; c < 4; ++c) nb[c] = t * 16 + c * 64;

            v8f acc[4] = {{}, {}, {}, {}};
            FragQ bc[4];
#pragma unroll
            for (int c = 0; c < 4; ++c) bc[c] = load_b(Wt, nb[c] + ln, 0, half);
#pragma unroll
            for (int kk = 0; kk < 7; ++kk) {
                FragQ bn[4];
#pragma unroll
                for (int c = 0; c < 4; ++c)
                    bn[c] = load_b(Wt, nb[c] + ln, (kk + 1) * 32, half);
#pragma unroll
                for (int c = 0; c < 4; ++c)
                    acc[c] = __builtin_amdgcn_wmma_f32_16x16x32_bf16(
                        false, a[kk].v, false, bc[c].v, (short)0, acc[c], false, false);
#pragma unroll
                for (int c = 0; c < 4; ++c) bc[c] = bn[c];
            }
#pragma unroll
            for (int c = 0; c < 4; ++c)
                acc[c] = __builtin_amdgcn_wmma_f32_16x16x32_bf16(
                    false, a[7].v, false, bc[c].v, (short)0, acc[c], false, false);

#pragma unroll
            for (int c = 0; c < 4; ++c) {
                float bias = b2[nb[c] + ln];
#pragma unroll
                for (int v = 0; v < 8; ++v) {
                    float r = fmaxf(acc[c][v] + bias, 0.0f);
                    atomicAdd(&sums[noff[v] + nb[c] + ln], r);
                }
            }
        }
    }
}

// ------------------------- finalize: mean ----------------------------------
__global__ void finalize_kernel(const float* __restrict__ sums,
                                const float* __restrict__ cnt,
                                float* __restrict__ out) {
    int i = blockIdx.x * blockDim.x + threadIdx.x;
    if (i < NODES * CH) {
        float c = cnt[i >> 8];
        out[i] = (c > 0.0f) ? sums[i] / c : 0.0f;
    }
}

// ---------------------------------------------------------------------------
extern "C" void kernel_launch(void* const* d_in, const int* in_sizes, int n_in,
                              void* d_out, int out_size, void* d_ws, size_t ws_size,
                              hipStream_t stream) {
    const float* X   = (const float*)d_in[0];   // [E, C]
    const int*   idx = (const int*)  d_in[1];   // [E]
    // d_in[2] = dim_size scalar (compile-time NODES)
    const float* W1  = (const float*)d_in[3];   // [C, C]
    const float* b1  = (const float*)d_in[4];   // [C]
    const float* W2  = (const float*)d_in[5];   // [C, C]
    const float* b2  = (const float*)d_in[6];   // [C]
    float* out = (float*)d_out;

    float* sums = (float*)d_ws;                 // [NODES, CH]
    float* cnt  = sums + (size_t)NODES * CH;    // [NODES]

    // 1) zero accumulators
    {
        int n = NODES * CH;
        zero_ws_kernel<<<(n + 255) / 256, 256, 0, stream>>>(sums, cnt);
    }
    // 2) per-node counts
    count_kernel<<<(EDGES + 255) / 256, 256, 0, stream>>>(idx, cnt);

    // 3) fused MLP + scatter-add
    {
        size_t smem = (size_t)(TE * RS + CH * RS + TE * RS) * sizeof(__bf16); // 264 KB
        dim3 grid(EDGES / TE);
        mlp_scatter_kernel<<<grid, NTHREADS, smem, stream>>>(X, idx, W1, b1, W2, b2, sums);
    }

    // 4) divide by counts (empty segments -> 0)
    {
        int n = NODES * CH;
        finalize_kernel<<<(n + 255) / 256, 256, 0, stream>>>(sums, cnt, out);
    }
}
